// RWKV7Attention_78391743086652
// MI455X (gfx1250) — compile-verified, matrix operand order checked by reading, not run on hardware
//
#include <hip/hip_runtime.h>
#include <hip/hip_bf16.h>
#include <math.h>

// ---------------------------------------------------------------------------
// RWKV7 attention forward for gfx1250 (MI455X).
// GEMMs use v_wmma_f32_16x16x32_bf16 (bf16 inputs, f32 accum) with
// fragment-major LDS tiles so every fragment loads as 2x ds_load_b128.
// Sequential delta-rule scan kept in f32 VGPR state (1 workgroup per (b,h)).
// ---------------------------------------------------------------------------

typedef __attribute__((ext_vector_type(16))) __bf16          v16bf;
typedef __attribute__((ext_vector_type(8)))  float           v8f;
typedef __attribute__((ext_vector_type(8)))  unsigned short  u16x8;
typedef __attribute__((ext_vector_type(4)))  float           f32x4;

struct bpair { u16x8 lo, hi; };   // 32 bytes == v16bf

#define DD   1024      // hidden size D
#define HH   16        // heads
#define DHD  64        // head dim

__device__ __forceinline__ unsigned short f2bf(float f) {
    unsigned int u = __builtin_bit_cast(unsigned int, f);
    u += 0x7FFFu + ((u >> 16) & 1u);          // round-to-nearest-even
    return (unsigned short)(u >> 16);
}
__device__ __forceinline__ float sigm(float x) { return 1.0f / (1.0f + expf(-x)); }

// 64-thread block reduction (2 waves) through LDS.
__device__ __forceinline__ float blk_reduce64(float v, float* s) {
    int tid = threadIdx.x;
    s[tid] = v;
    __syncthreads();
    for (int off = 32; off > 0; off >>= 1) {
        if (tid < off) s[tid] += s[tid + off];
        __syncthreads();
    }
    float r = s[0];
    __syncthreads();
    return r;
}

// ---------------------------------------------------------------------------
// f32 -> bf16(raw u16) conversion
// ---------------------------------------------------------------------------
__global__ void cvt_bf16_kernel(unsigned short* __restrict__ dst,
                                const float* __restrict__ src, int n) {
    int i = blockIdx.x * blockDim.x + threadIdx.x;
    if (i < n) dst[i] = f2bf(src[i]);
}

// ---------------------------------------------------------------------------
// activation (0=id, 1=tanh, 2=sigmoid) then f32 -> bf16
// ---------------------------------------------------------------------------
__global__ void act_bf16_kernel(unsigned short* __restrict__ dst,
                                const float* __restrict__ src, int n, int mode) {
    int i = blockIdx.x * blockDim.x + threadIdx.x;
    if (i >= n) return;
    float v = src[i];
    if (mode == 1) v = tanhf(v);
    else if (mode == 2) v = sigm(v);
    dst[i] = f2bf(v);
}

// ---------------------------------------------------------------------------
// Tiled bf16 WMMA GEMM:  C[M,N](f32) = A[M,K] @ B  (B: [K,N] or [N,K] if b_trans)
// a_mode==0: A is a raw-bf16 buffer Abf.
// a_mode==1: A[m,k] = x[m,k] + (shift(x)[m,k]-x[m,k])*mixv[k], cast to bf16
//            (token shift along T, zero row at t==0), K must equal D.
// Block: 128 threads (4 waves), C tile 64x64, K stepped by 32.
// LDS tiles are fragment-major (A: [m][k], B transposed: [n][k]) with an
// 80-byte row stride so each lane's fragment is 2 aligned 16-byte LDS loads.
// ---------------------------------------------------------------------------
__global__ void gemm_bf16_kernel(float* __restrict__ C,
                                 const unsigned short* __restrict__ Abf,
                                 const float* __restrict__ X,
                                 const float* __restrict__ mixv,
                                 const unsigned short* __restrict__ Bbf,
                                 int M, int N, int K,
                                 int a_mode, int b_trans, int T) {
    __shared__ unsigned short As[64][40];   // [m][k], stride 80B (16B-aligned)
    __shared__ unsigned short Bs[64][40];   // [n][k], transposed tile

    const int tid  = threadIdx.x;
    const int wave = tid >> 5;
    const int lane = tid & 31;
    const int hlf  = lane >> 4;      // 0/1 : lane half
    const int r16  = lane & 15;
    const int kb   = hlf * 8;

    const int m0 = blockIdx.y * 64;
    const int n0 = blockIdx.x * 64;

    v8f acc[4];
    const v8f vzero = {0.f, 0.f, 0.f, 0.f, 0.f, 0.f, 0.f, 0.f};
#pragma unroll
    for (int nt = 0; nt < 4; nt++) acc[nt] = vzero;

    for (int k0 = 0; k0 < K; k0 += 32) {
        // ---- stage A tile (64 rows x 32 k), 8 elements per chunk ----
#pragma unroll
        for (int i = 0; i < 2; i++) {
            int ch  = tid + i * 128;          // 0..255
            int row = ch >> 2;                // 0..63
            int c8  = (ch & 3) * 8;           // 0,8,16,24
            int m   = m0 + row;
            u16x8 pk;
            if (a_mode) {
                const float* xp = X + (long long)m * K + k0 + c8;
                f32x4 x0 = *(const f32x4*)xp;
                f32x4 x1 = *(const f32x4*)(xp + 4);
                f32x4 p0 = {0.f, 0.f, 0.f, 0.f};
                f32x4 p1 = {0.f, 0.f, 0.f, 0.f};
                if ((m % T) > 0) {
                    const float* pp = xp - K;
                    p0 = *(const f32x4*)pp;
                    p1 = *(const f32x4*)(pp + 4);
                }
                const float* mv = mixv + k0 + c8;
                f32x4 mv0 = *(const f32x4*)mv;
                f32x4 mv1 = *(const f32x4*)(mv + 4);
#pragma unroll
                for (int j = 0; j < 4; j++) {
                    pk[j]     = f2bf(x0[j] + (p0[j] - x0[j]) * mv0[j]);
                    pk[4 + j] = f2bf(x1[j] + (p1[j] - x1[j]) * mv1[j]);
                }
            } else {
                pk = *(const u16x8*)(Abf + (long long)m * K + k0 + c8);
            }
            *(u16x8*)&As[row][c8] = pk;
        }
        // ---- stage B tile transposed into Bs[n][k] ----
        if (b_trans) {
            // B is [N,K] row-major: contiguous along K, direct 16B copies
#pragma unroll
            for (int i = 0; i < 2; i++) {
                int ch = tid + i * 128;
                int n  = ch >> 2;
                int c8 = (ch & 3) * 8;
                u16x8 pk = {0, 0, 0, 0, 0, 0, 0, 0};
                if (n0 + n < N)
                    pk = *(const u16x8*)(Bbf + (long long)(n0 + n) * K + k0 + c8);
                *(u16x8*)&Bs[n][c8] = pk;
            }
        } else {
            // B is [K,N] row-major: wide global load along n, scatter into Bs
#pragma unroll
            for (int i = 0; i < 2; i++) {
                int ch  = tid + i * 128;
                int kk_ = ch >> 3;            // 0..31
                int n8  = (ch & 7) * 8;       // 0..56
                const unsigned short* bp = Bbf + (long long)(k0 + kk_) * N + n0 + n8;
                u16x8 pk = {0, 0, 0, 0, 0, 0, 0, 0};
                if (n0 + n8 + 7 < N) {
                    pk = *(const u16x8*)bp;
                } else {
#pragma unroll
                    for (int j = 0; j < 8; j++)
                        if (n0 + n8 + j < N) pk[j] = bp[j];
                }
#pragma unroll
                for (int j = 0; j < 8; j++) Bs[n8 + j][kk_] = pk[j];
            }
        }
        __syncthreads();

        // ---- A fragment: 2x aligned 16B LDS loads ----
        int mrow = wave * 16 + r16;
        bpair ap;
        ap.lo = *(const u16x8*)&As[mrow][kb];
        ap.hi = *(const u16x8*)&As[mrow][16 + kb];
        v16bf af = __builtin_bit_cast(v16bf, ap);

        // ---- 4 x WMMA across the 64-wide N tile ----
#pragma unroll
        for (int nt = 0; nt < 4; nt++) {
            int n = nt * 16 + r16;
            bpair bpr;
            bpr.lo = *(const u16x8*)&Bs[n][kb];
            bpr.hi = *(const u16x8*)&Bs[n][16 + kb];
            v16bf bfrag = __builtin_bit_cast(v16bf, bpr);
            acc[nt] = __builtin_amdgcn_wmma_f32_16x16x32_bf16(
                false, af, false, bfrag, (short)0, acc[nt], false, false);
        }
        __syncthreads();
    }

    // ---- store C (VGPR j: lanes 0-15 -> M=j, lanes 16-31 -> M=j+8) ----
#pragma unroll
    for (int nt = 0; nt < 4; nt++) {
        int n = n0 + nt * 16 + r16;
        if (n < N) {
#pragma unroll
            for (int j = 0; j < 8; j++) {
                int m = m0 + wave * 16 + hlf * 8 + j;
                C[(long long)m * N + n] = acc[nt][j];
            }
        }
    }
}

// ---------------------------------------------------------------------------
// Post kernel: per (token m, head h), 64 threads. In-place updates:
//   kk   = l2norm(k * k_k)              -> F_kk
//   a    = sigmoid(a0 + araw)           -> F_a
//   k    = k + k*(a-1)*k_a              -> F_k
//   w    = -e^{-0.5} * sigmoid(w0+wraw) -> F_w   (log-decay)
//   v    = v + (v_first - v)*sigmoid(v0+vlor) -> F_v
// ---------------------------------------------------------------------------
__global__ void post_kernel(float* __restrict__ F_k, float* __restrict__ F_a,
                            float* __restrict__ F_w, float* __restrict__ F_v,
                            const float* __restrict__ F_vl, float* __restrict__ F_kk,
                            const float* __restrict__ k_k, const float* __restrict__ k_a,
                            const float* __restrict__ a0, const float* __restrict__ w0,
                            const float* __restrict__ v0, const float* __restrict__ v_first) {
    __shared__ float sred[64];
    int mb = blockIdx.x;
    int m  = mb >> 4;          // token index
    int h  = mb & 15;          // head
    int tid = threadIdx.x;
    int c   = h * DHD + tid;   // channel
    long long idx = (long long)m * DD + c;

    float kraw  = F_k[idx];
    float kkpre = kraw * k_k[c];
    float ss    = blk_reduce64(kkpre * kkpre, sred);
    float kk    = kkpre / fmaxf(sqrtf(ss), 1e-12f);
    F_kk[idx]   = kk;

    float a = sigm(a0[c] + F_a[idx]);
    F_a[idx] = a;
    F_k[idx] = kraw + kraw * (a - 1.0f) * k_a[c];
    F_w[idx] = -0.6065306597126334f * sigm(w0[c] + F_w[idx]);  // -e^{-0.5}*sigmoid
    float vr = F_v[idx];
    F_v[idx] = vr + (v_first[idx] - vr) * sigm(v0[c] + F_vl[idx]);
}

// ---------------------------------------------------------------------------
// Sequential delta-rule scan. One block per (b,h), 64 threads; thread `tid`
// owns state row v=tid: S[v, k], k=0..63 held in 64 VGPRs.
//   S = S*diag(exp(w)) + (S@a) b^T + v k^T ;  o = S @ r ;  a=-kk, b=kk*icl
// ---------------------------------------------------------------------------
__global__ void scan_kernel(const float* __restrict__ R, const float* __restrict__ W,
                            const float* __restrict__ K, const float* __restrict__ V,
                            const float* __restrict__ KK, const float* __restrict__ A,
                            float* __restrict__ O, int T) {
    __shared__ float rv[64], dv[64], kv[64], vv[64], av[64], bv[64];
    int bh  = blockIdx.x;
    int b   = bh >> 4;
    int h   = bh & 15;
    int tid = threadIdx.x;

    float S[64];
#pragma unroll
    for (int i = 0; i < 64; i++) S[i] = 0.0f;

    for (int t = 0; t < T; t++) {
        long long idx = ((long long)(b * T + t)) * DD + h * DHD + tid;
        rv[tid] = R[idx];
        dv[tid] = expf(W[idx]);
        kv[tid] = K[idx];
        vv[tid] = V[idx];
        float kkv = KK[idx];
        av[tid] = -kkv;
        bv[tid] = kkv * A[idx];
        __syncthreads();

        float sa = 0.0f;
#pragma unroll
        for (int k = 0; k < 64; k++) sa += S[k] * av[k];

        float vval = vv[tid];
        float o = 0.0f;
#pragma unroll
        for (int k = 0; k < 64; k++) {
            float s = S[k] * dv[k] + sa * bv[k] + vval * kv[k];
            S[k] = s;
            o += s * rv[k];
        }
        O[idx] = o;
        __syncthreads();
    }
}

// ---------------------------------------------------------------------------
// GroupNorm (per head) + bonus + gate, write bf16 for final GEMM.
// ---------------------------------------------------------------------------
__global__ void norm_bonus_kernel(const float* __restrict__ F_o, const float* __restrict__ F_r,
                                  const float* __restrict__ F_k, const float* __restrict__ F_v,
                                  const float* __restrict__ F_g, const float* __restrict__ r_k,
                                  const float* __restrict__ gn_w, const float* __restrict__ gn_b,
                                  unsigned short* __restrict__ og) {
    __shared__ float sred[64];
    int mb = blockIdx.x;
    int m  = mb >> 4;
    int h  = mb & 15;
    int tid = threadIdx.x;
    int c   = h * DHD + tid;
    long long idx = (long long)m * DD + c;

    float o  = F_o[idx];
    float s1 = blk_reduce64(o, sred);
    float mu = s1 * (1.0f / 64.0f);
    float s2 = blk_reduce64(o * o, sred);
    float var = s2 * (1.0f / 64.0f) - mu * mu;
    float nrm = (o - mu) * rsqrtf(var + 6.4e-4f) * gn_w[c] + gn_b[c];  // eps = DH*1e-5

    float bn = blk_reduce64(F_r[idx] * F_k[idx] * r_k[c], sred);
    float outv = nrm + bn * F_v[idx];
    og[idx] = f2bf(outv * F_g[idx]);
}

// ---------------------------------------------------------------------------
// Host launcher
// ---------------------------------------------------------------------------
extern "C" void kernel_launch(void* const* d_in, const int* in_sizes, int n_in,
                              void* d_out, int out_size, void* d_ws, size_t ws_size,
                              hipStream_t stream) {
    // inputs in setup_inputs() order
    const float* hs      = (const float*)d_in[0];
    const float* v_first = (const float*)d_in[1];
    const float* x_r = (const float*)d_in[2];
    const float* x_w = (const float*)d_in[3];
    const float* x_k = (const float*)d_in[4];
    const float* x_v = (const float*)d_in[5];
    const float* x_a = (const float*)d_in[6];
    const float* x_g = (const float*)d_in[7];
    const float* w0  = (const float*)d_in[8];
    const float* w1  = (const float*)d_in[9];
    const float* w2  = (const float*)d_in[10];
    const float* a0  = (const float*)d_in[11];
    const float* a1  = (const float*)d_in[12];
    const float* a2  = (const float*)d_in[13];
    const float* v0  = (const float*)d_in[14];
    const float* v1  = (const float*)d_in[15];
    const float* v2  = (const float*)d_in[16];
    const float* g1  = (const float*)d_in[17];
    const float* g2  = (const float*)d_in[18];
    const float* k_k = (const float*)d_in[19];
    const float* k_a = (const float*)d_in[20];
    const float* r_k = (const float*)d_in[21];
    const float* w_r  = (const float*)d_in[22];
    const float* w_kp = (const float*)d_in[23];
    const float* w_vp = (const float*)d_in[24];
    const float* w_o  = (const float*)d_in[25];
    const float* gn_w = (const float*)d_in[26];
    const float* gn_b = (const float*)d_in[27];

    const int D  = 1024;
    const int M  = in_sizes[0] / D;   // B*T = 4096
    const int T  = 2048;
    const int Bb = M / T;
    const long long ND = (long long)M * D;

    // ---- bump allocator on d_ws ----
    char* wp = (char*)d_ws;
    auto alloc = [&](size_t bytes) -> void* {
        void* r = (void*)wp;
        wp += (bytes + 255) & ~(size_t)255;
        return r;
    };
    float* F_r   = (float*)alloc(ND * 4);
    float* F_w   = (float*)alloc(ND * 4);
    float* F_k   = (float*)alloc(ND * 4);
    float* F_v   = (float*)alloc(ND * 4);
    float* F_a   = (float*)alloc(ND * 4);
    float* F_vl  = (float*)alloc(ND * 4);
    float* F_kk  = (float*)alloc(ND * 4);
    float* F_g   = (float*)alloc(ND * 4);
    float* F_o   = (float*)alloc(ND * 4);
    float* F_mid = (float*)alloc((size_t)M * 160 * 4);
    unsigned short* mid_bf = (unsigned short*)alloc((size_t)M * 160 * 2);
    unsigned short* og_bf  = (unsigned short*)alloc(ND * 2);
    unsigned short* cWr  = (unsigned short*)alloc((size_t)D * D * 2);
    unsigned short* cWkp = (unsigned short*)alloc((size_t)D * D * 2);
    unsigned short* cWvp = (unsigned short*)alloc((size_t)D * D * 2);
    unsigned short* cWo  = (unsigned short*)alloc((size_t)D * D * 2);
    unsigned short* cW1 = (unsigned short*)alloc((size_t)D * 64 * 2);
    unsigned short* cW2 = (unsigned short*)alloc((size_t)64 * D * 2);
    unsigned short* cA1 = (unsigned short*)alloc((size_t)D * 64 * 2);
    unsigned short* cA2 = (unsigned short*)alloc((size_t)64 * D * 2);
    unsigned short* cV1 = (unsigned short*)alloc((size_t)D * 32 * 2);
    unsigned short* cV2 = (unsigned short*)alloc((size_t)32 * D * 2);
    unsigned short* cG1 = (unsigned short*)alloc((size_t)D * 160 * 2);
    unsigned short* cG2 = (unsigned short*)alloc((size_t)160 * D * 2);

    auto cvt = [&](unsigned short* dst, const float* src, int n) {
        cvt_bf16_kernel<<<dim3((n + 255) / 256), dim3(256), 0, stream>>>(dst, src, n);
    };
    cvt(cWr, w_r, D * D);  cvt(cWkp, w_kp, D * D);
    cvt(cWvp, w_vp, D * D); cvt(cWo, w_o, D * D);
    cvt(cW1, w1, D * 64);  cvt(cW2, w2, 64 * D);
    cvt(cA1, a1, D * 64);  cvt(cA2, a2, 64 * D);
    cvt(cV1, v1, D * 32);  cvt(cV2, v2, 32 * D);
    cvt(cG1, g1, D * 160); cvt(cG2, g2, 160 * D);

    auto gemm = [&](float* C, const unsigned short* Abf, const float* mixv,
                    const unsigned short* Bbf, int Mm, int Nn, int Kk,
                    int a_mode, int b_trans) {
        dim3 grid((Nn + 63) / 64, Mm / 64);
        gemm_bf16_kernel<<<grid, dim3(128), 0, stream>>>(
            C, Abf, hs, mixv, Bbf, Mm, Nn, Kk, a_mode, b_trans, T);
    };
    auto act = [&](unsigned short* dst, const float* src, int n, int mode) {
        act_bf16_kernel<<<dim3((n + 255) / 256), dim3(256), 0, stream>>>(dst, src, n, mode);
    };

    // main projections (A = token-shift-mixed hidden, fused into GEMM)
    gemm(F_r, nullptr, x_r, cWr,  M, D, D, 1, 1);   // r = xr @ w_r^T
    gemm(F_k, nullptr, x_k, cWkp, M, D, D, 1, 1);   // k = xk @ w_kp^T
    gemm(F_v, nullptr, x_v, cWvp, M, D, D, 1, 1);   // v = xv @ w_vp^T

    // w LoRA: tanh(xw @ w1) @ w2
    gemm(F_mid, nullptr, x_w, cW1, M, 64, D, 1, 0);
    act(mid_bf, F_mid, M * 64, 1);
    gemm(F_w, mid_bf, nullptr, cW2, M, D, 64, 0, 0);

    // a LoRA: (xa @ a1) @ a2
    gemm(F_mid, nullptr, x_a, cA1, M, 64, D, 1, 0);
    act(mid_bf, F_mid, M * 64, 0);
    gemm(F_a, mid_bf, nullptr, cA2, M, D, 64, 0, 0);

    // v LoRA: (xv @ v1) @ v2
    gemm(F_mid, nullptr, x_v, cV1, M, 32, D, 1, 0);
    act(mid_bf, F_mid, M * 32, 0);
    gemm(F_vl, mid_bf, nullptr, cV2, M, D, 32, 0, 0);

    // g LoRA: sigmoid(xg @ g1) @ g2
    gemm(F_mid, nullptr, x_g, cG1, M, 160, D, 1, 0);
    act(mid_bf, F_mid, M * 160, 2);
    gemm(F_g, mid_bf, nullptr, cG2, M, D, 160, 0, 0);

    // elementwise post-processing (kk / a / k / w / v residual)
    post_kernel<<<dim3(M * HH), dim3(64), 0, stream>>>(
        F_k, F_a, F_w, F_v, F_vl, F_kk, k_k, k_a, a0, w0, v0, v_first);

    // sequential delta-rule scan
    scan_kernel<<<dim3(Bb * HH), dim3(64), 0, stream>>>(
        F_r, F_w, F_k, F_v, F_kk, F_a, F_o, T);

    // group norm + bonus + gate -> bf16
    norm_bonus_kernel<<<dim3(M * HH), dim3(64), 0, stream>>>(
        F_o, F_r, F_k, F_v, F_g, r_k, gn_w, gn_b, og_bf);

    // output projection: out = (o*g) @ w_o^T
    gemm((float*)d_out, og_bf, nullptr, cWo, M, D, D, 0, 1);
}
